// _VehicleGNN_65274912965288
// MI455X (gfx1250) — compile-verified, matrix operand order checked by reading, not run on hardware
//
#include <hip/hip_runtime.h>
#include <hip/hip_bf16.h>

typedef __attribute__((ext_vector_type(16))) _Float16 v16h;
typedef __attribute__((ext_vector_type(8)))  _Float16 v8h;
typedef __attribute__((ext_vector_type(8)))  float    v8f;

#define WPB 8           // waves per block
#define VN  50000
#define EN  1600000
#define NVV 64

// Load one WMMA f16 operand (A-style) from an LDS tile stored row-major
// [row][stride] in halfs. 16-bit 16x32 operand layout (wave32):
// koff = 0 for lanes 0-15, 1 for lanes 16-31; K runs {kc*32+koff*8 ..+7} and +16.
__device__ __forceinline__ v16h load_op(const _Float16* tile, int stride,
                                        int row, int kc, int koff) {
    v16h a;
    const _Float16* p = tile + row * stride + kc * 32 + koff * 8;
    const _Float16* q = p + 16;
#pragma unroll
    for (int j = 0; j < 8; ++j) a[j] = p[j];
#pragma unroll
    for (int j = 0; j < 8; ++j) a[8 + j] = q[j];
    return a;
}

// A operand direct from a row-major f16 row in global memory: the two 8-half
// K-runs are 16B-contiguous -> two global_load_b128 per chunk.
__device__ __forceinline__ v16h load_ga(const _Float16* p) {
    v8h lo = *(const v8h*)(p);
    v8h hi = *(const v8h*)(p + 16);
    return __builtin_shufflevector(lo, hi, 0, 1, 2, 3, 4, 5, 6, 7,
                                   8, 9, 10, 11, 12, 13, 14, 15);
}

// ---------------------------------------------------------------- utilities
__global__ void zero_kernel(float* __restrict__ p, int n) {
    int i = blockIdx.x * blockDim.x + threadIdx.x;
    if (i < n) p[i] = 0.f;
}

__global__ void count_kernel(const int* __restrict__ dst, float* __restrict__ cnt, int E_) {
    int e = blockIdx.x * blockDim.x + threadIdx.x;
    if (e < E_) atomicAdd(&cnt[dst[e]], 1.0f);
}

// ---------------------------------------------------------------- props MLP (tiny)
__global__ void props_kernel(const float* __restrict__ vf,
                             const float* __restrict__ pW1, const float* __restrict__ pb1,
                             const float* __restrict__ pW2, const float* __restrict__ pb2,
                             float* __restrict__ props) {
    int nv = threadIdx.x;            // blockDim.x == 64
    float f[8];
#pragma unroll
    for (int i = 0; i < 8; ++i) f[i] = vf[nv * 8 + i];
    float hid[64];
#pragma unroll 4
    for (int d = 0; d < 64; ++d) {
        float s = pb1[d];
#pragma unroll
        for (int i = 0; i < 8; ++i) s += f[i] * pW1[i * 64 + d];
        hid[d] = s > 0.f ? s : 0.f;
    }
#pragma unroll 2
    for (int d2 = 0; d2 < 64; ++d2) {
        float s = pb2[d2];
#pragma unroll 8
        for (int d = 0; d < 64; ++d) s += hid[d] * pW2[d * 64 + d2];
        props[nv * 64 + d2] = s;
    }
}

// ---------------------------------------------------------------- node encoder
// h0 = concat(Z(64), meta(10)) @ nW(74x64) + nb  via WMMA (K 74->96);
// writes f32 h0 (residual path) and f16 copy (next layer's WMMA operand).
__global__ void node_enc_kernel(const float* __restrict__ Z,
                                const float* __restrict__ meta,
                                const float* __restrict__ nW,
                                const float* __restrict__ nb,
                                float* __restrict__ h0,
                                _Float16* __restrict__ h016, int V_) {
    __shared__ _Float16 sW[64 * 96];          // transposed [n][k], K padded to 96
    __shared__ float    sB[64];
    __shared__ _Float16 sTile[WPB][16 * 96];  // per-wave A tiles

    const int tid = threadIdx.x;
    for (int i = tid; i < 64 * 96; i += blockDim.x) {
        int n = i / 96, k = i % 96;
        sW[i] = (k < 74) ? (_Float16)nW[k * 64 + n] : (_Float16)0.f;
    }
    if (tid < 64) sB[tid] = nb[tid];
    __syncthreads();

    const int wave = tid >> 5, lane = tid & 31;
    const int ln = lane & 15, hi = lane >> 4;
    const int tile = blockIdx.x * WPB + wave;
    if (tile * 16 >= V_) return;              // wave-uniform

    _Float16* tA = &sTile[wave][0];
    const int v = tile * 16 + ln;
    if (hi == 0) {
        const float* zr = Z + (size_t)v * 64;
#pragma unroll
        for (int j = 0; j < 32; ++j) tA[ln * 96 + j] = (_Float16)zr[j];
    } else {
        const float* zr = Z + (size_t)v * 64 + 32;
#pragma unroll
        for (int j = 0; j < 32; ++j) tA[ln * 96 + 32 + j] = (_Float16)zr[j];
        const float* mr = meta + (size_t)v * 10;
#pragma unroll
        for (int j = 0; j < 10; ++j) tA[ln * 96 + 64 + j] = (_Float16)mr[j];
#pragma unroll
        for (int j = 74; j < 96; ++j) tA[ln * 96 + j] = (_Float16)0.f;
    }

    v8f acc[4];
#pragma unroll
    for (int nt = 0; nt < 4; ++nt)
#pragma unroll
        for (int j = 0; j < 8; ++j) acc[nt][j] = 0.f;

#pragma unroll
    for (int kc = 0; kc < 3; ++kc) {
        v16h a = load_op(tA, 96, ln, kc, hi);
#pragma unroll
        for (int nt = 0; nt < 4; ++nt) {
            v16h b = load_op(sW, 96, nt * 16 + ln, kc, hi);
            acc[nt] = __builtin_amdgcn_wmma_f32_16x16x32_f16(
                false, a, false, b, (short)0, acc[nt], false, false);
        }
    }
#pragma unroll
    for (int nt = 0; nt < 4; ++nt) {
        int n = nt * 16 + ln;
        float bb = sB[n];
#pragma unroll
        for (int r = 0; r < 8; ++r) {
            int m = r + hi * 8;
            size_t idx = (size_t)(tile * 16 + m) * 64 + n;
            float val = acc[nt][r] + bb;
            h0[idx] = val;
            h016[idx] = (_Float16)val;
        }
    }
}

// ---------------------------------------------------------------- edge conv core
// msg = relu(concat(h16[src],ea) @ W1 + b1) @ W2 + b2; atomic agg[dst] += msg.
// Persistent blocks; A operand for the h part loads straight from global f16
// rows (two b128 per K-chunk, L2-resident); GEMM2 B operands hoisted to VGPRs.
__global__ void edge_conv_kernel(const _Float16* __restrict__ h16,
                                 const float* __restrict__ edge_attr,
                                 const int* __restrict__ src,
                                 const int* __restrict__ dst,
                                 const float* __restrict__ W1, const float* __restrict__ b1,
                                 const float* __restrict__ W2, const float* __restrict__ b2,
                                 float* __restrict__ agg, int nTiles) {
    __shared__ _Float16 sW1[64 * 96];        // [n][k], K 80 padded to 96
    __shared__ _Float16 sW2[64 * 64];        // [n][k]
    __shared__ float    sB1[64], sB2[64];
    __shared__ _Float16 sMsg[WPB][16 * 64];  // relu(GEMM1) re-layout tile
    __shared__ int      sDst[WPB][16];

    const int tid = threadIdx.x;
    for (int i = tid; i < 64 * 96; i += blockDim.x) {
        int n = i / 96, k = i % 96;
        sW1[i] = (k < 80) ? (_Float16)W1[k * 64 + n] : (_Float16)0.f;
    }
    for (int i = tid; i < 64 * 64; i += blockDim.x) {
        int n = i >> 6, k = i & 63;
        sW2[i] = (_Float16)W2[k * 64 + n];
    }
    if (tid < 64) { sB1[tid] = b1[tid]; sB2[tid] = b2[tid]; }
    __syncthreads();

    const int wave = tid >> 5, lane = tid & 31;
    const int ln = lane & 15, hi = lane >> 4;
    _Float16* tM = &sMsg[wave][0];

    // loop-invariant GEMM2 B operands (8 x v16h = 64 VGPRs)
    v16h B2[2][4];
#pragma unroll
    for (int kc = 0; kc < 2; ++kc)
#pragma unroll
        for (int nt = 0; nt < 4; ++nt)
            B2[kc][nt] = load_op(sW2, 64, nt * 16 + ln, kc, hi);

    const int waveStride = gridDim.x * WPB;
    for (int tile = blockIdx.x * WPB + wave; tile < nTiles; tile += waveStride) {
        const int e = tile * 16 + ln;
        const int s = src[e];
        if (hi == 0) sDst[wave][ln] = dst[e];
        const _Float16* hrow = h16 + (size_t)s * 64;

        v8f acc[4];
#pragma unroll
        for (int nt = 0; nt < 4; ++nt)
#pragma unroll
            for (int j = 0; j < 8; ++j) acc[nt][j] = 0.f;

        // K-chunks 0,1: h features, A straight from global f16 row
#pragma unroll
        for (int kc = 0; kc < 2; ++kc) {
            v16h a = load_ga(hrow + kc * 32 + hi * 8);
#pragma unroll
            for (int nt = 0; nt < 4; ++nt) {
                v16h b = load_op(sW1, 96, nt * 16 + ln, kc, hi);
                acc[nt] = __builtin_amdgcn_wmma_f32_16x16x32_f16(
                    false, a, false, b, (short)0, acc[nt], false, false);
            }
        }
        // K-chunk 2: edge_attr (K 64..79) + zero pad (K 80..95)
        {
            v16h a;
            const float* ea = edge_attr + (size_t)e * 16 + hi * 8;
#pragma unroll
            for (int j = 0; j < 8; ++j) a[j] = (_Float16)ea[j];
#pragma unroll
            for (int j = 0; j < 8; ++j) a[8 + j] = (_Float16)0.f;
#pragma unroll
            for (int nt = 0; nt < 4; ++nt) {
                v16h b = load_op(sW1, 96, nt * 16 + ln, 2, hi);
                acc[nt] = __builtin_amdgcn_wmma_f32_16x16x32_f16(
                    false, a, false, b, (short)0, acc[nt], false, false);
            }
        }

        // bias + relu, re-layout C->A via per-wave LDS tile
#pragma unroll
        for (int nt = 0; nt < 4; ++nt) {
            int n = nt * 16 + ln;
            float bb = sB1[n];
#pragma unroll
            for (int r = 0; r < 8; ++r) {
                int m = r + hi * 8;
                float v = acc[nt][r] + bb;
                tM[m * 64 + n] = (_Float16)(v > 0.f ? v : 0.f);
            }
        }

        // GEMM2: [16x64] x [64x64], B from registers
        v8f acc2[4];
#pragma unroll
        for (int nt = 0; nt < 4; ++nt)
#pragma unroll
            for (int j = 0; j < 8; ++j) acc2[nt][j] = 0.f;
#pragma unroll
        for (int kc = 0; kc < 2; ++kc) {
            v16h a = load_op(tM, 64, ln, kc, hi);
#pragma unroll
            for (int nt = 0; nt < 4; ++nt) {
                acc2[nt] = __builtin_amdgcn_wmma_f32_16x16x32_f16(
                    false, a, false, B2[kc][nt], (short)0, acc2[nt], false, false);
            }
        }

        // scatter: agg[dst] atomics are L2-resident
        int dv[8];
#pragma unroll
        for (int r = 0; r < 8; ++r) dv[r] = sDst[wave][r + hi * 8];
#pragma unroll
        for (int nt = 0; nt < 4; ++nt) {
            int n = nt * 16 + ln;
            float bb = sB2[n];
#pragma unroll
            for (int r = 0; r < 8; ++r)
                atomicAdd(&agg[(size_t)dv[r] * 64 + n], acc2[nt][r] + bb);
        }
    }
}

// ---------------------------------------------------------------- residual + LN
// h_out = LN(h_in + agg/max(cnt,1)); emits f32 + f16 copies; re-zeroes agg.
__global__ void finalize_kernel(const float* __restrict__ h_in,
                                float* __restrict__ agg,
                                const float* __restrict__ cnt,
                                const float* __restrict__ g,
                                const float* __restrict__ be,
                                float* __restrict__ h_out,
                                _Float16* __restrict__ h_out16, int V_) {
    int gt = blockIdx.x * blockDim.x + threadIdx.x;
    int node = gt >> 5;
    int lane = threadIdx.x & 31;
    if (node >= V_) return;                   // wave-uniform
    size_t base = (size_t)node * 64;
    float c = cnt[node];
    float inv = 1.f / (c > 1.f ? c : 1.f);
    float x0 = h_in[base + lane]      + agg[base + lane]      * inv;
    float x1 = h_in[base + lane + 32] + agg[base + lane + 32] * inv;
    agg[base + lane] = 0.f;
    agg[base + lane + 32] = 0.f;
    float s = x0 + x1;
#pragma unroll
    for (int o = 16; o > 0; o >>= 1) s += __shfl_xor(s, o, 32);
    float mean = s * (1.f / 64.f);
    float d0 = x0 - mean, d1 = x1 - mean;
    float vv = d0 * d0 + d1 * d1;
#pragma unroll
    for (int o = 16; o > 0; o >>= 1) vv += __shfl_xor(vv, o, 32);
    float rstd = rsqrtf(vv * (1.f / 64.f) + 1e-5f);
    float y0 = d0 * rstd * g[lane]      + be[lane];
    float y1 = d1 * rstd * g[lane + 32] + be[lane + 32];
    h_out[base + lane]        = y0;
    h_out[base + lane + 32]   = y1;
    h_out16[base + lane]      = (_Float16)y0;
    h_out16[base + lane + 32] = (_Float16)y1;
}

// ---------------------------------------------------------------- output head
__global__ void out_kernel(const float* __restrict__ h,
                           const float* __restrict__ props,
                           const int* __restrict__ cur,
                           const float* __restrict__ rg,
                           const float* __restrict__ rb,
                           float* __restrict__ out, int NV_) {
    int gt = blockIdx.x * blockDim.x + threadIdx.x;
    int nv = gt >> 5;
    int lane = threadIdx.x & 31;
    if (nv >= NV_) return;
    size_t base = (size_t)cur[nv] * 64;
    float x0 = h[base + lane]      + props[nv * 64 + lane];
    float x1 = h[base + lane + 32] + props[nv * 64 + lane + 32];
    float s = x0 + x1;
#pragma unroll
    for (int o = 16; o > 0; o >>= 1) s += __shfl_xor(s, o, 32);
    float mean = s * (1.f / 64.f);
    float d0 = x0 - mean, d1 = x1 - mean;
    float vv = d0 * d0 + d1 * d1;
#pragma unroll
    for (int o = 16; o > 0; o >>= 1) vv += __shfl_xor(vv, o, 32);
    float rstd = rsqrtf(vv * (1.f / 64.f) + 1e-5f);
    out[nv * 64 + lane]      = d0 * rstd * rg[lane]      + rb[lane];
    out[nv * 64 + lane + 32] = d1 * rstd * rg[lane + 32] + rb[lane + 32];
}

// ---------------------------------------------------------------- launch
extern "C" void kernel_launch(void* const* d_in, const int* in_sizes, int n_in,
                              void* d_out, int out_size, void* d_ws, size_t ws_size,
                              hipStream_t stream) {
    const float* Z    = (const float*)d_in[0];
    const float* meta = (const float*)d_in[1];
    const float* vf   = (const float*)d_in[2];
    const float* ea   = (const float*)d_in[3];
    const int*   eidx = (const int*)d_in[4];
    const int*   cur  = (const int*)d_in[5];
    const float* nW   = (const float*)d_in[6];
    const float* nb   = (const float*)d_in[7];
    const float* pW1  = (const float*)d_in[8];
    const float* pb1  = (const float*)d_in[9];
    const float* pW2  = (const float*)d_in[10];
    const float* pb2  = (const float*)d_in[11];
    const float* rg   = (const float*)d_in[12];
    const float* rb   = (const float*)d_in[13];
    const float* l0W1 = (const float*)d_in[14];
    const float* l0b1 = (const float*)d_in[15];
    const float* l0W2 = (const float*)d_in[16];
    const float* l0b2 = (const float*)d_in[17];
    const float* l0g  = (const float*)d_in[18];
    const float* l0be = (const float*)d_in[19];
    const float* l1W1 = (const float*)d_in[20];
    const float* l1b1 = (const float*)d_in[21];
    const float* l1W2 = (const float*)d_in[22];
    const float* l1b2 = (const float*)d_in[23];
    const float* l1g  = (const float*)d_in[24];
    const float* l1be = (const float*)d_in[25];

    float* ws = (float*)d_ws;
    const size_t VD = (size_t)VN * 64;
    float* hEnc = ws;                  // [V,64] f32
    float* h1   = ws + VD;             // [V,64] f32
    float* aggB = ws + 2 * VD;         // [V,64] f32 (contiguous with cnt for zeroing)
    float* cntB = ws + 3 * VD;         // [V]
    float* prB  = cntB + VN;           // [NV,64]
    _Float16* hEnc16 = (_Float16*)(prB + (size_t)NVV * 64);  // [V,64] f16 (16B aligned)
    _Float16* h1_16  = hEnc16 + VD;                          // [V,64] f16

    const int* srcI = eidx;
    const int* dstI = eidx + EN;

    // 1) zero agg + cnt (contiguous)
    {
        int nz = (int)(VD + VN);
        zero_kernel<<<(nz + 255) / 256, 256, 0, stream>>>(aggB, nz);
    }
    // 2) in-degree counts
    count_kernel<<<(EN + 255) / 256, 256, 0, stream>>>(dstI, cntB, EN);
    // 3) vehicle props MLP
    props_kernel<<<1, 64, 0, stream>>>(vf, pW1, pb1, pW2, pb2, prB);
    // 4) node encoder (WMMA)
    {
        int tiles = (VN + 15) / 16;                // 3125
        int blocks = (tiles + WPB - 1) / WPB;      // 391
        node_enc_kernel<<<blocks, 32 * WPB, 0, stream>>>(Z, meta, nW, nb, hEnc, hEnc16, VN);
    }
    const int eTiles  = EN / 16;                   // 100000
    const int eBlocks = 1024;                      // persistent, grid-stride
    const int fBlocks = (VN * 32 + 255) / 256;     // one wave per node

    // 5) layer 0
    edge_conv_kernel<<<eBlocks, 32 * WPB, 0, stream>>>(hEnc16, ea, srcI, dstI,
                                                       l0W1, l0b1, l0W2, l0b2, aggB, eTiles);
    finalize_kernel<<<fBlocks, 256, 0, stream>>>(hEnc, aggB, cntB, l0g, l0be, h1, h1_16, VN);
    // 6) layer 1 (agg re-zeroed by finalize above)
    edge_conv_kernel<<<eBlocks, 32 * WPB, 0, stream>>>(h1_16, ea, srcI, dstI,
                                                       l1W1, l1b1, l1W2, l1b2, aggB, eTiles);
    finalize_kernel<<<fBlocks, 256, 0, stream>>>(h1, aggB, cntB, l1g, l1be, hEnc, hEnc16, VN);
    // 7) output head
    out_kernel<<<(NVV * 32 + 255) / 256, 256, 0, stream>>>(hEnc, prB, cur, rg, rb,
                                                           (float*)d_out, NVV);
}
